// keops_matern_kernel_88734024335764
// MI455X (gfx1250) — matrix operand order repaired and checked by reading, not compiled, not run on hardware
//
#include <hip/hip_runtime.h>
#include <stdint.h>

// ---- problem constants (from reference) ----
constexpr int cN  = 16384;
constexpr int cM  = 16384;
constexpr int cD  = 8;
constexpr int cDV = 16;
constexpr int JSPLIT = 4;

__device__ __constant__ const float C1 = 2.23606797749978969f;   // sqrt(5)
__device__ __constant__ const float C2 = 1.29099444873580563f;   // sqrt(5/3)

typedef __attribute__((ext_vector_type(16))) _Float16 v16h;
typedef __attribute__((ext_vector_type(8)))  float    v8f;
typedef __attribute__((ext_vector_type(2)))  float    v2f;

// ---------------------------------------------------------------------------
// Prep 1: per point,
//   sx[i] = sum_d ls_d*x[i,d]^2        (resp. sy[j])
//   xp[i] = x row permuted [d0,d1,d4,d5 | d2,d3,d6,d7]       (B-operand order)
//   yp[j] = (ls*y) row, same permutation                     (A-operand order)
// so each lane of the main kernel fetches its 4 Gram floats as one b128.
// ---------------------------------------------------------------------------
__global__ __launch_bounds__(256) void prep_points(const float* __restrict__ x,
                                                   const float* __restrict__ y,
                                                   const float* __restrict__ ls,
                                                   float* __restrict__ sx,
                                                   float* __restrict__ sy,
                                                   float* __restrict__ xp,
                                                   float* __restrict__ yp) {
    int idx = blockIdx.x * blockDim.x + threadIdx.x;
    float l[cD];
#pragma unroll
    for (int d = 0; d < cD; ++d) l[d] = ls[d];
    const int perm[cD] = {0, 1, 4, 5, 2, 3, 6, 7};
    if (idx < cN) {
        const float* r = x + (size_t)idx * cD;
        float v[cD];
#pragma unroll
        for (int d = 0; d < cD; ++d) v[d] = r[d];
        float s = 0.f;
#pragma unroll
        for (int d = 0; d < cD; ++d) s = fmaf(l[d] * v[d], v[d], s);
        sx[idx] = s;
        float* o = xp + (size_t)idx * cD;
#pragma unroll
        for (int d = 0; d < cD; ++d) o[d] = v[perm[d]];          // unscaled
    } else if (idx < cN + cM) {
        const int j = idx - cN;
        const float* r = y + (size_t)j * cD;
        float v[cD];
#pragma unroll
        for (int d = 0; d < cD; ++d) v[d] = r[d];
        float s = 0.f;
#pragma unroll
        for (int d = 0; d < cD; ++d) s = fmaf(l[d] * v[d], v[d], s);
        sy[j] = s;
        float* o = yp + (size_t)j * cD;
#pragma unroll
        for (int d = 0; d < cD; ++d) o[d] = l[perm[d]] * v[perm[d]];  // ls-scaled
    }
}

// ---------------------------------------------------------------------------
// Prep 2: pack b into f16 in the exact V_WMMA_F32_16X16X32_F16 B-operand
// layout, grouped so each lane's 8 dwords per 32-j chunk are contiguous:
//   bpk[(((chunk*2 + hi)*16 + dv)*8) + v] = pack_f16(b[j][dv], b[j+1][dv])
//   with j = chunk*32 + hi*8 + 2v + (v >= 4 ? 8 : 0)
// ---------------------------------------------------------------------------
__global__ __launch_bounds__(256) void prep_b(const float* __restrict__ b,
                                              uint32_t* __restrict__ bpk) {
    int idx = blockIdx.x * blockDim.x + threadIdx.x;
    const int total = (cM / 32) * 2 * 16 * 8;
    if (idx >= total) return;
    int v  = idx & 7;
    int dv = (idx >> 3) & 15;
    int hi = (idx >> 7) & 1;
    int c  = idx >> 8;
    int j  = c * 32 + hi * 8 + 2 * v + (v >= 4 ? 8 : 0);
    union { _Float16 h[2]; uint32_t u; } p;
    p.h[0] = (_Float16)b[(size_t)j * cDV + dv];
    p.h[1] = (_Float16)b[(size_t)(j + 1) * cDV + dv];
    bpk[idx] = p.u;
}

// ---------------------------------------------------------------------------
// Main: one wave = one 16-row i-tile x one j-segment.
//  - G'[j,i] = (ls*y) @ x^T via 2x V_WMMA_F32_16X16X4_F32   (f32 precision)
//  - Matern elementwise on VALU/trans pipes
//  - out += K @ b via V_WMMA_F32_16X16X32_F16 (K values land directly in the
//    f16 A-operand layout: i = lane%16, j = r + 8*(lane>=16) (+16 per subtile))
// All broadcast/operand data arrives as aligned global_load_b128.
// ---------------------------------------------------------------------------
__global__ __launch_bounds__(256) void matern_main(const float* __restrict__ xp,
                                                   const float* __restrict__ yp,
                                                   const float* __restrict__ sx,
                                                   const float* __restrict__ sy,
                                                   const uint32_t* __restrict__ bpk,
                                                   float* __restrict__ outbase,
                                                   int jseg) {
    const int lane  = threadIdx.x & 31;
    const int lo    = lane & 15;          // i_local / dv column
    const int hi    = lane >> 4;          // K-half selector
    const int itile = blockIdx.x * 8 + (threadIdx.x >> 5);
    const int i0    = itile * 16;
    const int seg   = blockIdx.y;

    // B-operand of the Gram WMMA: permuted x row, one b128 (i = lane%16)
    const float4 xq = *(const float4*)(xp + (size_t)(i0 + lo) * cD + hi * 4);
    v2f bx0; bx0.x = xq.x; bx0.y = xq.y;      // dims {2h, 2h+1}
    v2f bx1; bx1.x = xq.z; bx1.y = xq.w;      // dims {4+2h, 5+2h}
    const float sxi = sx[i0 + lo];

    v8f acc = (v8f)0.0f;   // out tile accumulator [i = r+8*hi, dv = lane%16]

    const int jbeg = seg * jseg;
    const int jend = jbeg + jseg;
    for (int j0 = jbeg; j0 < jend; j0 += 32) {
        v16h A;   // f16 kernel values, A-operand of 16x16x32 WMMA
#pragma unroll
        for (int t = 0; t < 2; ++t) {
            const int jt = j0 + 16 * t;
            // A-operand of Gram WMMA: prescaled+permuted y row, one b128
            const float4 yq = *(const float4*)(yp + (size_t)(jt + lo) * cD + hi * 4);
            v2f ay0; ay0.x = yq.x; ay0.y = yq.y;
            v2f ay1; ay1.x = yq.z; ay1.y = yq.w;
            v8f g = (v8f)0.0f;
            g = __builtin_amdgcn_wmma_f32_16x16x4_f32(false, ay0, false, bx0,
                                                      (short)0, g, false, false);
            g = __builtin_amdgcn_wmma_f32_16x16x4_f32(false, ay1, false, bx1,
                                                      (short)0, g, false, false);
            // g[r] = G'[j = jt + r + 8*hi, i = i0 + lane%16]
            // sy broadcast: 8 consecutive floats, two aligned b128 loads
            const float4 s0 = *(const float4*)(sy + jt + 8 * hi);
            const float4 s1 = *(const float4*)(sy + jt + 8 * hi + 4);
            const float syv[8] = {s0.x, s0.y, s0.z, s0.w, s1.x, s1.y, s1.z, s1.w};
#pragma unroll
            for (int r = 0; r < 8; ++r) {
                const float d2 = fmaxf(syv[r] + sxi - 2.0f * g[r], 0.0f);
                const float rr = __builtin_amdgcn_sqrtf(d2);
                const float e  = __expf(-C1 * rr);
                const float kv = fmaf(C1, rr, fmaf(C2, d2, 1.0f)) * e;
                A[8 * t + r] = (_Float16)kv;   // lands exactly in A layout
            }
        }
        // B-operand: prepacked f16 b tile, two 16B loads per lane
        const uint32_t* bb = bpk + (size_t)((((j0 >> 5) * 2 + hi) * 16 + lo) * 8);
        union { uint4 q[2]; v16h h; } ub;
        ub.q[0] = *(const uint4*)(bb);
        ub.q[1] = *(const uint4*)(bb + 4);
        acc = __builtin_amdgcn_wmma_f32_16x16x32_f16(false, A, false, ub.h,
                                                     (short)0, acc, false, false);
    }

    // D layout: out[i = r + 8*hi, dv = lane%16]
    float* pout = outbase + (size_t)seg * cN * cDV + (size_t)i0 * cDV;
#pragma unroll
    for (int r = 0; r < 8; ++r)
        pout[(r + 8 * hi) * cDV + lo] = acc[r];
}

// ---------------------------------------------------------------------------
// Reduce JSPLIT partial buffers -> out (deterministic, no atomics)
// ---------------------------------------------------------------------------
__global__ __launch_bounds__(256) void reduce_out(const float* __restrict__ partial,
                                                  float* __restrict__ out) {
    int idx = blockIdx.x * blockDim.x + threadIdx.x;
    if (idx >= cN * cDV) return;
    float s = 0.f;
#pragma unroll
    for (int t = 0; t < JSPLIT; ++t) s += partial[(size_t)t * cN * cDV + idx];
    out[idx] = s;
}

extern "C" void kernel_launch(void* const* d_in, const int* in_sizes, int n_in,
                              void* d_out, int out_size, void* d_ws, size_t ws_size,
                              hipStream_t stream) {
    const float* ls = (const float*)d_in[0];
    const float* x  = (const float*)d_in[1];
    const float* y  = (const float*)d_in[2];
    const float* b  = (const float*)d_in[3];
    float* out = (float*)d_out;

    char* ws = (char*)d_ws;
    float*    sx  = (float*)(ws);                        //  64 KB
    float*    sy  = (float*)(ws + (64 << 10));           //  64 KB
    uint32_t* bpk = (uint32_t*)(ws + (128 << 10));       // 512 KB
    float*    xp  = (float*)(ws + (640 << 10));          // 512 KB (permuted x)
    float*    yp  = (float*)(ws + (1152 << 10));         // 512 KB (scaled+permuted y)
    float*    par = (float*)(ws + (1664 << 10));         //   4 MB (JSPLIT partials)

    prep_points<<<(cN + cM + 255) / 256, 256, 0, stream>>>(x, y, ls, sx, sy, xp, yp);
    const int bpk_dw = (cM / 32) * 2 * 16 * 8;
    prep_b<<<(bpk_dw + 255) / 256, 256, 0, stream>>>(b, bpk);

    const size_t need_split = ((size_t)1664 << 10) + (size_t)JSPLIT * cN * cDV * sizeof(float);
    dim3 block(256);
    if (ws_size >= need_split) {
        dim3 grid(cN / 16 / 8, JSPLIT);                  // 128 x 4 blocks, 4096 waves
        matern_main<<<grid, block, 0, stream>>>(xp, yp, sx, sy, bpk, par, cM / JSPLIT);
        reduce_out<<<(cN * cDV + 255) / 256, 256, 0, stream>>>(par, out);
    } else {
        dim3 grid(cN / 16 / 8, 1);                       // direct write, no partials
        matern_main<<<grid, block, 0, stream>>>(xp, yp, sx, sy, bpk, out, cM);
    }
}
